// HGCN_68281390071888
// MI455X (gfx1250) — matrix operand order repaired and checked by reading, not compiled, not run on hardware
//
#include <hip/hip_runtime.h>
#include <hip/hip_bf16.h>
#include <cstddef>

#define NPTS 131072
#define M1G  512
#define GS1C 256
#define M2G  64
#define KNN  20
#define EPSB 1e-5f

typedef __attribute__((ext_vector_type(16))) __bf16 v16bf;
typedef __attribute__((ext_vector_type(8)))  float  v8f;

union FragB { unsigned int u[8]; v16bf v; };

__device__ __forceinline__ float lrelu(float x, float s) { return x >= 0.f ? x : x * s; }

__device__ __forceinline__ unsigned short f2bf(float f) {
  unsigned int u = __float_as_uint(f);
  unsigned int r = u + 0x7FFFu + ((u >> 16) & 1u);
  return (unsigned short)(r >> 16);
}

// ---------------------------------------------------------------------------
// Fold BN into classifier weights, convert to bf16 (runs once, tiny).
// W1f: 128x224, W2f: 64x128, W3f: 16x64 (rows 13..15 zero-padded).
// ---------------------------------------------------------------------------
__global__ __launch_bounds__(256) void fold_kernel(
    const float* __restrict__ w1, const float* __restrict__ g1,
    const float* __restrict__ bb1, const float* __restrict__ mm1,
    const float* __restrict__ vv1, const float* __restrict__ w2,
    const float* __restrict__ g2, const float* __restrict__ bb2,
    const float* __restrict__ mm2, const float* __restrict__ vv2,
    const float* __restrict__ w3, unsigned short* __restrict__ W1f,
    float* __restrict__ B1, unsigned short* __restrict__ W2f,
    float* __restrict__ B2, unsigned short* __restrict__ W3f) {
  int t = blockIdx.x * blockDim.x + threadIdx.x;
  int stride = gridDim.x * blockDim.x;
  for (int i = t; i < 128 * 224; i += stride) {
    int o = i / 224;
    float s = g1[o] * rsqrtf(vv1[o] + EPSB);
    W1f[i] = f2bf(w1[i] * s);
  }
  for (int i = t; i < 128; i += stride)
    B1[i] = bb1[i] - g1[i] * rsqrtf(vv1[i] + EPSB) * mm1[i];
  for (int i = t; i < 64 * 128; i += stride) {
    int o = i >> 7;
    float s = g2[o] * rsqrtf(vv2[o] + EPSB);
    W2f[i] = f2bf(w2[i] * s);
  }
  for (int i = t; i < 64; i += stride)
    B2[i] = bb2[i] - g2[i] * rsqrtf(vv2[i] + EPSB) * mm2[i];
  for (int i = t; i < 16 * 64; i += stride) {
    int o = i >> 6;
    W3f[i] = (o < 13) ? f2bf(w3[i]) : (unsigned short)0;
  }
}

// ---------------------------------------------------------------------------
// Phase 1: per-point local MLP -> pe (bf16, [64][N]), per-group max -> ge1,
// nearest-centroid argmins m1 (512 centers) and m2 (64 centers).
// One block per group of 256 points.
// ---------------------------------------------------------------------------
__global__ __launch_bounds__(256) void phase1_kernel(
    const float* __restrict__ X, const float* __restrict__ coords1,
    const float* __restrict__ origins1, const float* __restrict__ coords2,
    const float* __restrict__ wl1g, const float* __restrict__ wl2g,
    unsigned short* __restrict__ peB, float* __restrict__ ge1,
    int* __restrict__ m1, int* __restrict__ m2) {
  __shared__ float c1[3 * 512];
  __shared__ float c2[3 * 64];
  __shared__ float wl1[36];
  __shared__ float wl2[384];
  __shared__ float red[8 * 64];
  int t = threadIdx.x, g = blockIdx.x;
  for (int i = t; i < 1536; i += 256) c1[i] = coords1[i];
  for (int i = t; i < 192; i += 256) c2[i] = coords2[i];
  if (t < 36) wl1[t] = wl1g[t];
  for (int i = t; i < 384; i += 256) wl2[i] = wl2g[i];
  __syncthreads();

  size_t p = (size_t)g * GS1C + t;
  float xv[6];
#pragma unroll
  for (int c = 0; c < 6; ++c) xv[c] = X[(size_t)c * NPTS + p];
  float r0 = xv[0], r1 = xv[1], r2 = xv[2];   // raw coords (pre-shift)
  xv[0] -= origins1[g * 3 + 0];
  xv[1] -= origins1[g * 3 + 1];
  xv[2] -= origins1[g * 3 + 2];

  float f6[6];
#pragma unroll
  for (int o = 0; o < 6; ++o) {
    float s = 0.f;
#pragma unroll
    for (int c = 0; c < 6; ++c) s += wl1[o * 6 + c] * xv[c];
    f6[o] = lrelu(s, 0.2f);
  }
  float pe[64];
#pragma unroll
  for (int o = 0; o < 64; ++o) {
    float s = 0.f;
#pragma unroll
    for (int c = 0; c < 6; ++c) s += wl2[o * 6 + c] * f6[c];
    pe[o] = lrelu(s, 0.2f);
    peB[(size_t)o * NPTS + p] = f2bf(pe[o]);
  }

  // group max -> ge1[64][512]
  int lane = t & 31, wav = t >> 5;
  for (int o = 0; o < 64; ++o) {
    float v = pe[o];
    for (int off = 16; off; off >>= 1) v = fmaxf(v, __shfl_xor(v, off, 32));
    if (lane == 0) red[wav * 64 + o] = v;
  }
  __syncthreads();
  if (t < 64) {
    float mx = red[t];
#pragma unroll
    for (int wv = 1; wv < 8; ++wv) mx = fmaxf(mx, red[wv * 64 + t]);
    ge1[t * M1G + g] = mx;
  }

  // argmin of |c|^2 - 2 r.c  (same argmin as full squared distance)
  float bd = 3.4e38f; int bi = 0;
  for (int j = 0; j < 512; ++j) {
    float cx = c1[j], cy = c1[512 + j], cz = c1[1024 + j];
    float d = cx * cx + cy * cy + cz * cz - 2.f * (r0 * cx + r1 * cy + r2 * cz);
    if (d < bd) { bd = d; bi = j; }
  }
  m1[p] = bi;
  bd = 3.4e38f; bi = 0;
  for (int j = 0; j < 64; ++j) {
    float cx = c2[j], cy = c2[64 + j], cz = c2[128 + j];
    float d = cx * cx + cy * cy + cz * cz - 2.f * (r0 * cx + r1 * cy + r2 * cz);
    if (d < bd) { bd = d; bi = j; }
  }
  m2[p] = bi;
}

// ---------------------------------------------------------------------------
// top-K (K=20) of pairwise pd among M points (M = 512 or 64). One block,
// M threads; ties broken toward lower index (matches jax.lax.top_k).
// ---------------------------------------------------------------------------
__global__ __launch_bounds__(512) void knn_kernel(const float* __restrict__ coords,
                                                  int M, int* __restrict__ idx) {
  __shared__ float c[3 * 512];
  int t = threadIdx.x;
  for (int i = t; i < 3 * M; i += blockDim.x) c[i] = coords[i];
  __syncthreads();
  if (t >= M) return;
  float x0 = c[t], y0 = c[M + t], z0 = c[2 * M + t];
  float n0 = x0 * x0 + y0 * y0 + z0 * z0;
  float prevv = 3.4e38f; int previ = -1;
  for (int s = 0; s < KNN; ++s) {
    float bestv = -3.4e38f; int besti = 0;
    for (int j = 0; j < M; ++j) {
      float xj = c[j], yj = c[M + j], zj = c[2 * M + j];
      float pd = 2.f * (x0 * xj + y0 * yj + z0 * zj) - n0 -
                 (xj * xj + yj * yj + zj * zj);
      bool elig = (pd < prevv) || (pd == prevv && j > previ);
      if (elig && pd > bestv) { bestv = pd; besti = j; }
    }
    idx[t * KNN + s] = besti;
    prevv = bestv; previ = besti;
  }
}

// ---------------------------------------------------------------------------
// graph_embed stage A: x1[o][m] = mean_k lrelu(bn(w1 @ [ge ; rel]))
// f channels: [0..Cf) = ge (k-independent), [Cf..Cf+3) = coord diffs.
// grid = M blocks, blockDim = O threads.
// ---------------------------------------------------------------------------
__global__ __launch_bounds__(128) void gembedA_kernel(
    int M, int Cf, const float* __restrict__ coords, const float* __restrict__ ge,
    const float* __restrict__ w1, const float* __restrict__ bg,
    const float* __restrict__ bb, const float* __restrict__ bm,
    const float* __restrict__ bv, const int* __restrict__ idx,
    float* __restrict__ x1out) {
  __shared__ float gecol[128];
  __shared__ int nb[KNN];
  int m = blockIdx.x, o = threadIdx.x;
  for (int i = o; i < Cf; i += blockDim.x) gecol[i] = ge[i * M + m];
  if (o < KNN) nb[o] = idx[m * KNN + o];
  __syncthreads();
  int rl = Cf + 3;
  float base = 0.f;
  for (int c = 0; c < Cf; ++c) base += w1[o * rl + c] * gecol[c];
  float wa = w1[o * rl + Cf], wbc = w1[o * rl + Cf + 1], wc = w1[o * rl + Cf + 2];
  float cm0 = coords[m], cm1 = coords[M + m], cm2 = coords[2 * M + m];
  float s = bg[o] * rsqrtf(bv[o] + EPSB);
  float bia = bb[o] - s * bm[o];
  float acc = 0.f;
  for (int k = 0; k < KNN; ++k) {
    int j = nb[k];
    float d0 = coords[j] - cm0, d1 = coords[M + j] - cm1, d2 = coords[2 * M + j] - cm2;
    float v = base + wa * d0 + wbc * d1 + wc * d2;
    acc += lrelu(s * v + bia, 0.2f);
  }
  x1out[o * M + m] = acc * (1.f / KNN);
}

// ---------------------------------------------------------------------------
// graph_embed stage B: h[o][m] = mean_k lrelu(bn(w2 @ [diff ; center]))
// w2 row layout: [0..C)=diff weights, [C..2C)=center weights; O == C.
// ---------------------------------------------------------------------------
__global__ __launch_bounds__(128) void gembedB_kernel(
    int M, int C, const float* __restrict__ x1, const float* __restrict__ w2,
    const float* __restrict__ bg, const float* __restrict__ bb,
    const float* __restrict__ bm, const float* __restrict__ bv,
    const int* __restrict__ idx, float* __restrict__ hout) {
  __shared__ float xc[96];
  __shared__ int nb[KNN];
  int m = blockIdx.x, o = threadIdx.x;
  xc[o] = x1[o * M + m];
  if (o < KNN) nb[o] = idx[m * KNN + o];
  __syncthreads();
  float cen = 0.f, dbase = 0.f;
  for (int c = 0; c < C; ++c) {
    cen += w2[o * 2 * C + C + c] * xc[c];
    dbase += w2[o * 2 * C + c] * xc[c];
  }
  float s = bg[o] * rsqrtf(bv[o] + EPSB);
  float bia = bb[o] - s * bm[o];
  float acc = 0.f;
  for (int k = 0; k < KNN; ++k) {
    int j = nb[k];
    float dsum = 0.f;
    for (int c = 0; c < C; ++c) dsum += w2[o * 2 * C + c] * x1[c * M + j];
    float v = cen + (dsum - dbase);
    acc += lrelu(s * v + bia, 0.2f);
  }
  hout[o * M + m] = acc * (1.f / KNN);
}

// ---------------------------------------------------------------------------
// h2 local MLPs (64->64->128) over the 512 columns of h1, then max over
// groups of 8 -> ge2[128][64]. One block per group, 128 threads.
// ---------------------------------------------------------------------------
__global__ __launch_bounds__(128) void local2_kernel(
    const float* __restrict__ h1, const float* __restrict__ wA,
    const float* __restrict__ wB, float* __restrict__ ge2) {
  __shared__ float col[64];
  __shared__ float fcol[64];
  int gi = blockIdx.x, t = threadIdx.x;
  float mx = -3.4e38f;
  for (int s = 0; s < 8; ++s) {
    int m = gi * 8 + s;
    if (t < 64) col[t] = h1[t * M1G + m];
    __syncthreads();
    if (t < 64) {
      float a = 0.f;
      for (int c = 0; c < 64; ++c) a += wA[t * 64 + c] * col[c];
      fcol[t] = lrelu(a, 0.2f);
    }
    __syncthreads();
    float b = 0.f;
    for (int c = 0; c < 64; ++c) b += wB[t * 64 + c] * fcol[c];
    mx = fmaxf(mx, lrelu(b, 0.2f));
    __syncthreads();
  }
  ge2[t * M2G + gi] = mx;
}

// ---------------------------------------------------------------------------
// Fused WMMA classifier: per 16-point tile build cat[224] (pe | h1-gather |
// h2-gather) in LDS as bf16, then chain 224->128->64->13 with
// v_wmma_f32_16x16x32_bf16, bias+lrelu between layers through LDS.
// 256 threads = 8 waves; wave w owns output rows [16w,16w+16).
// ---------------------------------------------------------------------------
#define CATS 116   // ushort stride for cat[n][k], k in [0,224)
#define Y1S  68    // ushort stride for y1[n][m], m in [0,128)
#define Y2S  36    // ushort stride for y2[n][m], m in [0,64)

__global__ __launch_bounds__(256) void cls_kernel(
    const unsigned short* __restrict__ peB, const float* __restrict__ h1,
    const float* __restrict__ h2, const int* __restrict__ m1,
    const int* __restrict__ m2, const unsigned short* __restrict__ W1f,
    const float* __restrict__ B1, const unsigned short* __restrict__ W2f,
    const float* __restrict__ B2, const unsigned short* __restrict__ W3f,
    float* __restrict__ out) {
  __shared__ unsigned short cat[16 * CATS];
  __shared__ unsigned short y1[16 * Y1S];
  __shared__ unsigned short y2[16 * Y2S];
  __shared__ int sm1[16], sm2[16];
  int t = threadIdx.x;
  size_t base = (size_t)blockIdx.x * 16;
  if (t < 16) { sm1[t] = m1[base + t]; sm2[t] = m2[base + t]; }
  __syncthreads();

  for (int e = t; e < 224 * 16; e += 256) {
    int k = e >> 4, n = e & 15;
    unsigned short val;
    if (k < 64)       val = peB[(size_t)k * NPTS + base + n];
    else if (k < 128) val = f2bf(h1[(k - 64) * M1G + sm1[n]]);
    else              val = f2bf(h2[(k - 128) * M2G + sm2[n]]);
    cat[n * CATS + k] = val;
  }
  __syncthreads();

  int wave = t >> 5, lane = t & 31;
  int h = lane >> 4, ln = lane & 15;

  // ---- layer 1: 128 x 224 @ 224 x 16 ----
  {
    v8f acc = {0.f, 0.f, 0.f, 0.f, 0.f, 0.f, 0.f, 0.f};
    int mrow = wave * 16 + ln;
    for (int ks = 0; ks < 7; ++ks) {
      FragB A, Bf;
#pragma unroll
      for (int v = 0; v < 8; ++v) {
        int ka = 32 * ks + 16 * h + 2 * v;
        A.u[v] = *reinterpret_cast<const unsigned int*>(W1f + mrow * 224 + ka);
        int kb = 32 * ks + 16 * h + 2 * v;
        Bf.u[v] = *reinterpret_cast<const unsigned int*>(&cat[ln * CATS + kb]);
      }
      acc = __builtin_amdgcn_wmma_f32_16x16x32_bf16(false, A.v, false, Bf.v,
                                                    (short)0, acc, false, false);
    }
#pragma unroll
    for (int r = 0; r < 8; r += 2) {
      int m0 = wave * 16 + 8 * h + r;
      float a0 = lrelu(acc[r] + B1[m0], 0.1f);
      float a1 = lrelu(acc[r + 1] + B1[m0 + 1], 0.1f);
      unsigned int pk = (unsigned int)f2bf(a0) | ((unsigned int)f2bf(a1) << 16);
      *reinterpret_cast<unsigned int*>(&y1[ln * Y1S + m0]) = pk;
    }
  }
  __syncthreads();

  // ---- layer 2: 64 x 128 @ 128 x 16 (waves 0..3) ----
  if (wave < 4) {
    v8f acc = {0.f, 0.f, 0.f, 0.f, 0.f, 0.f, 0.f, 0.f};
    int mrow = wave * 16 + ln;
    for (int ks = 0; ks < 4; ++ks) {
      FragB A, Bf;
#pragma unroll
      for (int v = 0; v < 8; ++v) {
        int ka = 32 * ks + 16 * h + 2 * v;
        A.u[v] = *reinterpret_cast<const unsigned int*>(W2f + mrow * 128 + ka);
        int kb = 32 * ks + 16 * h + 2 * v;
        Bf.u[v] = *reinterpret_cast<const unsigned int*>(&y1[ln * Y1S + kb]);
      }
      acc = __builtin_amdgcn_wmma_f32_16x16x32_bf16(false, A.v, false, Bf.v,
                                                    (short)0, acc, false, false);
    }
#pragma unroll
    for (int r = 0; r < 8; r += 2) {
      int m0 = wave * 16 + 8 * h + r;
      float a0 = lrelu(acc[r] + B2[m0], 0.1f);
      float a1 = lrelu(acc[r + 1] + B2[m0 + 1], 0.1f);
      unsigned int pk = (unsigned int)f2bf(a0) | ((unsigned int)f2bf(a1) << 16);
      *reinterpret_cast<unsigned int*>(&y2[ln * Y2S + m0]) = pk;
    }
  }
  __syncthreads();

  // ---- layer 3: 16(13) x 64 @ 64 x 16 (wave 0), fp32 out ----
  if (wave == 0) {
    v8f acc = {0.f, 0.f, 0.f, 0.f, 0.f, 0.f, 0.f, 0.f};
    int mrow = ln;
    for (int ks = 0; ks < 2; ++ks) {
      FragB A, Bf;
#pragma unroll
      for (int v = 0; v < 8; ++v) {
        int ka = 32 * ks + 16 * h + 2 * v;
        A.u[v] = *reinterpret_cast<const unsigned int*>(W3f + mrow * 64 + ka);
        int kb = 32 * ks + 16 * h + 2 * v;
        Bf.u[v] = *reinterpret_cast<const unsigned int*>(&y2[ln * Y2S + kb]);
      }
      acc = __builtin_amdgcn_wmma_f32_16x16x32_bf16(false, A.v, false, Bf.v,
                                                    (short)0, acc, false, false);
    }
#pragma unroll
    for (int r = 0; r < 8; ++r) {
      int m0 = 8 * h + r;
      if (m0 < 13) out[(size_t)m0 * NPTS + base + ln] = acc[r];
    }
  }
}

// ---------------------------------------------------------------------------
// Host-side launch. Inputs flattened leaf-wise in setup_inputs() dict order:
//  0:X 1:coords1 2:origins1 3:coords2 4:h1_local_w1 5:h1_local_w2 6:h1_g_w1
//  7..10:h1_g_bn1{g,b,m,v} 11:h1_g_w2 12..15:h1_g_bn2 16:h2_local_w1
//  17:h2_local_w2 18:h2_g_w1 19..22:h2_g_bn1 23:h2_g_w2 24..27:h2_g_bn2
//  28:cls_w1 29..32:cls_bn1 33:cls_w2 34..37:cls_bn2 38:cls_w3
// ---------------------------------------------------------------------------
extern "C" void kernel_launch(void* const* d_in, const int* in_sizes, int n_in,
                              void* d_out, int out_size, void* d_ws, size_t ws_size,
                              hipStream_t stream) {
  (void)in_sizes; (void)n_in; (void)out_size; (void)ws_size;
  const float* X        = (const float*)d_in[0];
  const float* coords1  = (const float*)d_in[1];
  const float* origins1 = (const float*)d_in[2];
  const float* coords2  = (const float*)d_in[3];
  const float* h1l_w1   = (const float*)d_in[4];
  const float* h1l_w2   = (const float*)d_in[5];
  const float* h1g_w1   = (const float*)d_in[6];
  const float* h1g_bn1g = (const float*)d_in[7];
  const float* h1g_bn1b = (const float*)d_in[8];
  const float* h1g_bn1m = (const float*)d_in[9];
  const float* h1g_bn1v = (const float*)d_in[10];
  const float* h1g_w2   = (const float*)d_in[11];
  const float* h1g_bn2g = (const float*)d_in[12];
  const float* h1g_bn2b = (const float*)d_in[13];
  const float* h1g_bn2m = (const float*)d_in[14];
  const float* h1g_bn2v = (const float*)d_in[15];
  const float* h2l_w1   = (const float*)d_in[16];
  const float* h2l_w2   = (const float*)d_in[17];
  const float* h2g_w1   = (const float*)d_in[18];
  const float* h2g_bn1g = (const float*)d_in[19];
  const float* h2g_bn1b = (const float*)d_in[20];
  const float* h2g_bn1m = (const float*)d_in[21];
  const float* h2g_bn1v = (const float*)d_in[22];
  const float* h2g_w2   = (const float*)d_in[23];
  const float* h2g_bn2g = (const float*)d_in[24];
  const float* h2g_bn2b = (const float*)d_in[25];
  const float* h2g_bn2m = (const float*)d_in[26];
  const float* h2g_bn2v = (const float*)d_in[27];
  const float* cls_w1   = (const float*)d_in[28];
  const float* cls_bn1g = (const float*)d_in[29];
  const float* cls_bn1b = (const float*)d_in[30];
  const float* cls_bn1m = (const float*)d_in[31];
  const float* cls_bn1v = (const float*)d_in[32];
  const float* cls_w2   = (const float*)d_in[33];
  const float* cls_bn2g = (const float*)d_in[34];
  const float* cls_bn2b = (const float*)d_in[35];
  const float* cls_bn2m = (const float*)d_in[36];
  const float* cls_bn2v = (const float*)d_in[37];
  const float* cls_w3   = (const float*)d_in[38];
  float* out = (float*)d_out;

  // workspace carve-up (256B aligned)
  char* wp = (char*)d_ws;
  auto carve = [&](size_t bytes) {
    char* p = wp;
    wp += (bytes + 255) & ~(size_t)255;
    return p;
  };
  unsigned short* peB = (unsigned short*)carve((size_t)64 * NPTS * 2);
  int*   m1v  = (int*)carve((size_t)NPTS * 4);
  int*   m2v  = (int*)carve((size_t)NPTS * 4);
  float* ge1  = (float*)carve(64 * 512 * 4);
  int*   idx1 = (int*)carve(512 * KNN * 4);
  float* x1b  = (float*)carve(64 * 512 * 4);
  float* h1b  = (float*)carve(64 * 512 * 4);
  float* ge2  = (float*)carve(128 * 64 * 4);
  int*   idx2 = (int*)carve(64 * KNN * 4);
  float* x2b  = (float*)carve(96 * 64 * 4);
  float* h2b  = (float*)carve(96 * 64 * 4);
  unsigned short* W1f = (unsigned short*)carve(128 * 224 * 2);
  float* B1 = (float*)carve(128 * 4);
  unsigned short* W2f = (unsigned short*)carve(64 * 128 * 2);
  float* B2 = (float*)carve(64 * 4);
  unsigned short* W3f = (unsigned short*)carve(16 * 64 * 2);

  fold_kernel<<<64, 256, 0, stream>>>(cls_w1, cls_bn1g, cls_bn1b, cls_bn1m,
                                      cls_bn1v, cls_w2, cls_bn2g, cls_bn2b,
                                      cls_bn2m, cls_bn2v, cls_w3, W1f, B1, W2f,
                                      B2, W3f);
  phase1_kernel<<<512, 256, 0, stream>>>(X, coords1, origins1, coords2, h1l_w1,
                                         h1l_w2, peB, ge1, m1v, m2v);
  knn_kernel<<<1, 512, 0, stream>>>(coords1, 512, idx1);
  gembedA_kernel<<<512, 64, 0, stream>>>(512, 64, coords1, ge1, h1g_w1,
                                         h1g_bn1g, h1g_bn1b, h1g_bn1m,
                                         h1g_bn1v, idx1, x1b);
  gembedB_kernel<<<512, 64, 0, stream>>>(512, 64, x1b, h1g_w2, h1g_bn2g,
                                         h1g_bn2b, h1g_bn2m, h1g_bn2v, idx1,
                                         h1b);
  local2_kernel<<<64, 128, 0, stream>>>(h1b, h2l_w1, h2l_w2, ge2);
  knn_kernel<<<1, 64, 0, stream>>>(coords2, 64, idx2);
  gembedA_kernel<<<64, 96, 0, stream>>>(64, 128, coords2, ge2, h2g_w1,
                                        h2g_bn1g, h2g_bn1b, h2g_bn1m, h2g_bn1v,
                                        idx2, x2b);
  gembedB_kernel<<<64, 96, 0, stream>>>(64, 96, x2b, h2g_w2, h2g_bn2g,
                                        h2g_bn2b, h2g_bn2m, h2g_bn2v, idx2,
                                        h2b);
  cls_kernel<<<NPTS / 16, 256, 0, stream>>>(peB, h1b, h2b, m1v, m2v, W1f, B1,
                                            W2f, B2, W3f, out);
}
